// CrossAttentionBlock_87926570484475
// MI455X (gfx1250) — matrix operand order repaired and checked by reading, not compiled, not run on hardware
//
#include <hip/hip_runtime.h>
#include <hip/hip_bf16.h>
#include <hip/hip_fp16.h>

typedef __attribute__((ext_vector_type(16))) _Float16 v16h;
typedef __attribute__((ext_vector_type(8)))  float    v8f;

#define BB 2
#define CT 64
#define CO 32
#define EE 64
#define HI 32
#define WI 32
#define HOq 96
#define WOq 96
#define NN (HOq*WOq)   /* 9216 */

union FragH { v16h v; unsigned u[8]; };

__device__ __forceinline__ unsigned lds_off_of(const void* p) {
  // generic pointer to __shared__: low 32 bits are the LDS byte address
  return (unsigned)(unsigned long long)p;
}

__device__ __forceinline__ void async_b128(unsigned lds, const void* g) {
  asm volatile("global_load_async_to_lds_b128 %0, %1, off"
               :: "v"(lds), "v"(g) : "memory");
}

__device__ __forceinline__ void pack8(FragH& f, uint4 lo, uint4 hi) {
  f.u[0]=lo.x; f.u[1]=lo.y; f.u[2]=lo.z; f.u[3]=lo.w;
  f.u[4]=hi.x; f.u[5]=hi.y; f.u[6]=hi.z; f.u[7]=hi.w;
}

// ---- DPP16 butterfly reduction within each 16-lane half (no LDS traffic) ----
// step pairs: quad xor1, quad xor2, half-row mirror (8-groups), row mirror (16)
template<int CTRL>
__device__ __forceinline__ float dppf(float x) {
  return __int_as_float(
      __builtin_amdgcn_update_dpp(0, __float_as_int(x), CTRL, 0xf, 0xf, true));
}
__device__ __forceinline__ float redmax16(float x) {
  x = fmaxf(x, dppf<0xB1>(x));    // quad_perm [1,0,3,2]
  x = fmaxf(x, dppf<0x4E>(x));    // quad_perm [2,3,0,1]
  x = fmaxf(x, dppf<0x141>(x));   // row_half_mirror
  x = fmaxf(x, dppf<0x140>(x));   // row_mirror
  return x;
}
__device__ __forceinline__ float redsum16(float x) {
  x += dppf<0xB1>(x);
  x += dppf<0x4E>(x);
  x += dppf<0x141>(x);
  x += dppf<0x140>(x);
  return x;
}

// ---------------------------------------------------------------------------
// Kernel 1: bilinear 3x upsample (half-pixel) + Q = q_w*x_up + q_b, scaled by
// 1/sqrt(E)=0.125, stored f16 row-major [b][n][e].
// ---------------------------------------------------------------------------
__global__ void qproj_kernel(const float* __restrict__ xt,
                             const float* __restrict__ qw,
                             const float* __restrict__ qb,
                             _Float16* __restrict__ Qh) {
  __shared__ float sw[EE*CT];
  __shared__ float sb[EE];
  for (int i = threadIdx.x; i < EE*CT; i += blockDim.x) sw[i] = qw[i];
  if (threadIdx.x < EE) sb[threadIdx.x] = qb[threadIdx.x];
  __syncthreads();
  int tid = blockIdx.x*blockDim.x + threadIdx.x;
  if (tid >= BB*NN) return;
  int b = tid / NN, n = tid % NN;
  int ho = n / WOq, wo = n % WOq;
  float fy = (ho + 0.5f)*(1.0f/3.0f) - 0.5f;
  float fx = (wo + 0.5f)*(1.0f/3.0f) - 0.5f;
  float y0f = floorf(fy), x0f = floorf(fx);
  float wy = fy - y0f, wx = fx - x0f;
  int y0 = (int)y0f, x0 = (int)x0f;
  int y1 = y0 + 1, x1 = x0 + 1;
  y0 = min(max(y0,0),HI-1); y1 = min(max(y1,0),HI-1);
  x0 = min(max(x0,0),WI-1); x1 = min(max(x1,0),WI-1);
  float xup[CT];
  const float* xb = xt + (size_t)b*CT*HI*WI;
  #pragma unroll 4
  for (int c = 0; c < CT; ++c) {
    const float* p = xb + c*HI*WI;
    float a00 = p[y0*WI+x0], a01 = p[y0*WI+x1];
    float a10 = p[y1*WI+x0], a11 = p[y1*WI+x1];
    float top = a00 + wx*(a01 - a00);
    float bot = a10 + wx*(a11 - a10);
    xup[c] = top + wy*(bot - top);
  }
  _Float16* qo = Qh + (size_t)(b*NN + n)*EE;
  for (int e = 0; e < EE; ++e) {
    float acc = sb[e];
    const float* wr = sw + e*CT;
    #pragma unroll 8
    for (int c = 0; c < CT; ++c) acc += wr[c]*xup[c];
    qo[e] = (_Float16)(acc * 0.125f);
  }
}

// ---------------------------------------------------------------------------
// Kernel 2: K = k_w*x_opt + k_b stored f16 [b][m][e];
//           V = v_w*x_opt + v_b stored f16 TRANSPOSED [b][o][m].
// ---------------------------------------------------------------------------
__global__ void kvproj_kernel(const float* __restrict__ xo,
                              const float* __restrict__ kw,
                              const float* __restrict__ kb,
                              const float* __restrict__ vw,
                              const float* __restrict__ vb,
                              _Float16* __restrict__ Kh,
                              _Float16* __restrict__ Vh) {
  __shared__ float skw[EE*CO];
  __shared__ float svw[CO*CO];
  __shared__ float skb[EE], svb[CO];
  for (int i = threadIdx.x; i < EE*CO; i += blockDim.x) skw[i] = kw[i];
  for (int i = threadIdx.x; i < CO*CO; i += blockDim.x) svw[i] = vw[i];
  if (threadIdx.x < EE) skb[threadIdx.x] = kb[threadIdx.x];
  if (threadIdx.x < CO) svb[threadIdx.x] = vb[threadIdx.x];
  __syncthreads();
  int tid = blockIdx.x*blockDim.x + threadIdx.x;
  if (tid >= BB*NN) return;
  int b = tid / NN, n = tid % NN;
  float xi[CO];
  const float* xp = xo + (size_t)b*CO*NN + n;
  #pragma unroll
  for (int c = 0; c < CO; ++c) xi[c] = xp[(size_t)c*NN];
  _Float16* ko = Kh + (size_t)(b*NN + n)*EE;
  for (int e = 0; e < EE; ++e) {
    float acc = skb[e];
    const float* wr = skw + e*CO;
    #pragma unroll
    for (int c = 0; c < CO; ++c) acc += wr[c]*xi[c];
    ko[e] = (_Float16)acc;
  }
  for (int o = 0; o < CO; ++o) {
    float acc = svb[o];
    const float* wr = svw + o*CO;
    #pragma unroll
    for (int c = 0; c < CO; ++c) acc += wr[c]*xi[c];
    Vh[(size_t)(b*CO + o)*NN + n] = (_Float16)acc;
  }
}

// ---------------------------------------------------------------------------
// Kernel 3: flash attention; 8 waves/block share double-buffered async-staged
// K/V tiles. 1 wave = 16 query rows. DPP16 butterfly softmax reductions.
// ---------------------------------------------------------------------------
__global__ void __launch_bounds__(256,1)
attn_kernel(const _Float16* __restrict__ Qh,
            const _Float16* __restrict__ Kh,
            const _Float16* __restrict__ Vh,
            float* __restrict__ Obuf) {
  __shared__ __align__(16) _Float16 Ktile[2][64][64];  // 2 x 8KB
  __shared__ __align__(16) _Float16 Vtile[2][32][64];  // 2 x 4KB  [o][m]
  __shared__ __align__(16) _Float16 Plds[8][16][64];   // 16KB, per-wave P relay

  const int tid  = threadIdx.x;
  const int wv   = tid >> 5;
  const int lane = tid & 31;
  const int col  = lane & 15;
  const int grp  = lane >> 4;
  const int b    = blockIdx.x / (NN/128);
  const int tile = blockIdx.x % (NN/128);
  const int n0   = tile*128 + wv*16;

  const _Float16* Kbase = Kh + (size_t)b*NN*EE;
  const _Float16* Vbase = Vh + (size_t)b*CO*NN;

  // per-thread async staging: K block = 512 x 16B (contiguous in global+LDS),
  // V block = 256 x 16B (row-contiguous). 3 transfers/thread/iteration.
  auto stage = [&](int buf, int m0) {
    unsigned kdst = lds_off_of(&Ktile[buf][0][0]);
    const _Float16* ksrc = Kbase + (size_t)m0*EE;
    #pragma unroll
    for (int k2 = 0; k2 < 2; ++k2) {
      int j = tid + 256*k2;
      async_b128(kdst + j*16, ksrc + j*8);
    }
    {
      int j = tid;
      int o = j >> 3, seg = j & 7;
      async_b128(lds_off_of(&Vtile[buf][o][0]) + seg*16,
                 Vbase + (size_t)o*NN + m0 + seg*8);
    }
  };

  // resident Q A-fragments (e-chunks of 32)
  FragH qa[2];
  {
    const uint4* qp = (const uint4*)(Qh + (size_t)(b*NN + n0 + col)*EE);
    #pragma unroll
    for (int ec = 0; ec < 2; ++ec)
      pack8(qa[ec], qp[ec*4 + grp], qp[ec*4 + 2 + grp]);
  }

  v8f oacc[2]; oacc[0] = 0.f; oacc[1] = 0.f;
  float mrun[8], lrun[8];
  #pragma unroll
  for (int r = 0; r < 8; ++r) { mrun[r] = -3.0e38f; lrun[r] = 0.f; }

  stage(0, 0);

  const int NIT = NN/64;   // 144
  for (int it = 0; it < NIT; ++it) {
    const int cur = it & 1, nxt = cur ^ 1;
    const int mpre = (it + 1 < NIT) ? (it + 1)*64 : 0;
    stage(nxt, mpre);                                 // prefetch next block
    asm volatile("s_wait_asynccnt 0x3" ::: "memory"); // drain current block
    __syncthreads();                                  // visible block-wide

    // ---- S = Q*K^T over 4 sixteen-column subtiles (K from LDS) ----
    v8f s[4];
    #pragma unroll
    for (int st = 0; st < 4; ++st) {
      v8f acc = 0.f;
      const uint4* kp = (const uint4*)(&Ktile[cur][st*16 + col][0]);
      #pragma unroll
      for (int ec = 0; ec < 2; ++ec) {
        FragH kf;
        pack8(kf, kp[ec*4 + grp], kp[ec*4 + 2 + grp]);
        acc = __builtin_amdgcn_wmma_f32_16x16x32_f16(
                false, qa[ec].v, false, kf.v, (short)0, acc, false, false);
      }
      s[st] = acc;
    }

    // ---- online softmax (row stats per C-layout register/half) ----
    float fac[8];
    #pragma unroll
    for (int r = 0; r < 8; ++r) {
      float vmx = fmaxf(fmaxf(s[0][r], s[1][r]), fmaxf(s[2][r], s[3][r]));
      vmx = redmax16(vmx);
      float mn = fmaxf(mrun[r], vmx);
      fac[r] = __expf(mrun[r] - mn);
      mrun[r] = mn;
      float rs = 0.f;
      #pragma unroll
      for (int st = 0; st < 4; ++st) {
        float p = __expf(s[st][r] - mn);
        s[st][r] = p;
        rs += p;
      }
      rs = redsum16(rs);
      lrun[r] = lrun[r]*fac[r] + rs;
    }
    #pragma unroll
    for (int ot = 0; ot < 2; ++ot)
      #pragma unroll
      for (int r = 0; r < 8; ++r) oacc[ot][r] *= fac[r];

    // ---- P (C-layout) -> LDS -> A-layout fragments ----
    #pragma unroll
    for (int st = 0; st < 4; ++st)
      #pragma unroll
      for (int r = 0; r < 8; ++r)
        Plds[wv][r + 8*grp][st*16 + col] = (_Float16)s[st][r];
    asm volatile("s_wait_dscnt 0" ::: "memory");

    // ---- O += P * V^T (V from LDS) ----
    const uint4* pp = (const uint4*)(&Plds[wv][col][0]);
    #pragma unroll
    for (int mc = 0; mc < 2; ++mc) {
      FragH pa;
      pack8(pa, pp[mc*4 + grp], pp[mc*4 + 2 + grp]);
      #pragma unroll
      for (int ot = 0; ot < 2; ++ot) {
        FragH vf;
        const uint4* vp = (const uint4*)(&Vtile[cur][ot*16 + col][0]);
        pack8(vf, vp[mc*4 + grp], vp[mc*4 + 2 + grp]);
        oacc[ot] = __builtin_amdgcn_wmma_f32_16x16x32_f16(
                     false, pa.v, false, vf.v, (short)0, oacc[ot], false, false);
      }
    }
    __syncthreads();   // all waves done with buffer `cur`; safe to overwrite
  }

  // ---- finalize: divide by softmax denom, store O [b][n][o] f32 ----
  #pragma unroll
  for (int r = 0; r < 8; ++r) {
    float inv = 1.0f / lrun[r];
    int row = n0 + r + 8*grp;
    float* op = Obuf + (size_t)(b*NN + row)*CO;
    op[col]      = oacc[0][r]*inv;
    op[16 + col] = oacc[1][r]*inv;
  }
}

// ---------------------------------------------------------------------------
// Kernel 4: g = out_w * O, BN(eval) folded to scale/offset, ReLU.
// ---------------------------------------------------------------------------
__global__ void epilogue_kernel(const float* __restrict__ Obuf,
                                const float* __restrict__ ow,
                                const float* __restrict__ gamma,
                                const float* __restrict__ beta,
                                const float* __restrict__ mean,
                                const float* __restrict__ var,
                                float* __restrict__ out) {
  __shared__ float sw[CT*CO];
  __shared__ float ssc[CT], sof[CT];
  for (int i = threadIdx.x; i < CT*CO; i += blockDim.x) sw[i] = ow[i];
  if (threadIdx.x < CT) {
    int t = threadIdx.x;
    float sc = gamma[t] * rsqrtf(var[t] + 1e-5f);
    ssc[t] = sc;
    sof[t] = beta[t] - mean[t]*sc;
  }
  __syncthreads();
  int tid = blockIdx.x*blockDim.x + threadIdx.x;
  if (tid >= BB*NN) return;
  int b = tid / NN, n = tid % NN;
  float ov[CO];
  const float* op = Obuf + (size_t)(b*NN + n)*CO;
  #pragma unroll
  for (int o = 0; o < CO; ++o) ov[o] = op[o];
  for (int t = 0; t < CT; ++t) {
    float acc = 0.f;
    const float* wr = sw + t*CO;
    #pragma unroll
    for (int o = 0; o < CO; ++o) acc += wr[o]*ov[o];
    acc = acc*ssc[t] + sof[t];
    out[(size_t)(b*CT + t)*NN + n] = fmaxf(acc, 0.f);
  }
}

// ---------------------------------------------------------------------------
extern "C" void kernel_launch(void* const* d_in, const int* in_sizes, int n_in,
                              void* d_out, int out_size, void* d_ws, size_t ws_size,
                              hipStream_t stream) {
  const float* xt = (const float*)d_in[0];
  const float* xo = (const float*)d_in[1];
  const float* qw = (const float*)d_in[2];
  const float* qb = (const float*)d_in[3];
  const float* kw = (const float*)d_in[4];
  const float* kb = (const float*)d_in[5];
  const float* vw = (const float*)d_in[6];
  const float* vb = (const float*)d_in[7];
  const float* ow = (const float*)d_in[8];
  const float* gm = (const float*)d_in[9];
  const float* be = (const float*)d_in[10];
  const float* mn = (const float*)d_in[11];
  const float* vr = (const float*)d_in[12];

  char* ws = (char*)d_ws;
  const size_t QH_BYTES = (size_t)BB*NN*EE*2;   // 2,359,296
  const size_t VH_BYTES = (size_t)BB*NN*CO*2;   // 1,179,648
  _Float16* Qh = (_Float16*)(ws);
  _Float16* Kh = (_Float16*)(ws + QH_BYTES);
  _Float16* Vh = (_Float16*)(ws + 2*QH_BYTES);
  float*    Ob = (float*)   (ws + 2*QH_BYTES + VH_BYTES);

  const int nthr = BB*NN;                       // 18432 = 72 * 256
  qproj_kernel  <<<nthr/256, 256, 0, stream>>>(xt, qw, qb, Qh);
  kvproj_kernel <<<nthr/256, 256, 0, stream>>>(xo, kw, kb, vw, vb, Kh, Vh);
  attn_kernel   <<<BB*(NN/128), 256, 0, stream>>>(Qh, Kh, Vh, Ob);
  epilogue_kernel<<<nthr/256, 256, 0, stream>>>(Ob, ow, gm, be, mn, vr, (float*)d_out);
}
